// ImageLinearSelfAttention_58609123721902
// MI455X (gfx1250) — matrix operand order repaired and checked by reading, not compiled, not run on hardware
//
#include <hip/hip_runtime.h>
#include <hip/hip_bf16.h>

typedef __attribute__((ext_vector_type(16))) _Float16 v16h;
typedef __attribute__((ext_vector_type(8)))  _Float16 v8h;
typedef __attribute__((ext_vector_type(8)))  float    v8f;
typedef __attribute__((vector_size(16)))     int      v4i;   // matches builtin param

#define B_    4
#define C_    32
#define N_    1024   /* H*W */
#define D_    1024   /* per-head channel dim = C*DIM_HEAD */
#define HID   4096
#define HEADS 4
#define BH    16     /* B*HEADS */

#define KSTEP 32
#define MT    128    /* workgroup M tile */
#define NTL   256    /* workgroup N tile */

#if __has_builtin(__builtin_amdgcn_global_load_async_to_lds_b128) && \
    __has_builtin(__builtin_amdgcn_s_wait_asynccnt)
#define USE_ASYNC 1
#else
#define USE_ASYNC 0
#endif

#define AS_GLOBAL __attribute__((address_space(1)))
#define AS_LOCAL  __attribute__((address_space(3)))

union AF { v16h v; v8h h2[2]; _Float16 e[16]; };
union HF { v8h  v; _Float16 e[8]; };
union DF { v8f  v; float e[8]; };

__device__ inline v8f wmma16(v16h a, v16h b, v8f c) {
  return __builtin_amdgcn_wmma_f32_16x16x32_f16(false, a, false, b, (short)0, c,
                                                false, false);
}

// 16-byte global -> LDS copy; async (ASYNCcnt) when available.
__device__ inline void async_cp16(const _Float16* g, _Float16* l) {
#if USE_ASYNC
  __builtin_amdgcn_global_load_async_to_lds_b128(
      (AS_GLOBAL v4i*)g, (AS_LOCAL v4i*)l, 0, 0);
#else
  *(v8h*)l = *(const v8h*)g;
#endif
}
__device__ inline void async_wait0() {
#if USE_ASYNC
  __builtin_amdgcn_s_wait_asynccnt(0);
#endif
}

// A fragment: 16(M)x32(K) f16, row-major source, leading dim ld (elements).
// Lane L: m=L&15, g=L>>4; halves 0..7 <- k=8g.., halves 8..15 <- k=16+8g..
__device__ inline v16h ldfrag_a(const _Float16* p, int ld) {
  int lane = threadIdx.x & 31;
  int m = lane & 15, g = lane >> 4;
  const _Float16* r = p + (size_t)m * ld;
  AF f;
  f.h2[0] = *(const v8h*)(r + 8 * g);
  f.h2[1] = *(const v8h*)(r + 16 + 8 * g);
  return f.v;
}

// B fragment: 32(K)x16(N) whose column n is row n of a row-major [N,K] matrix.
// Lane L: n=L&15, g=L>>4; halves 0..7 <- k=8g.., halves 8..15 <- k=16+8g..
__device__ inline v16h ldfrag_bt(const _Float16* p, int ld) {
  int lane = threadIdx.x & 31;
  int n = lane & 15, g = lane >> 4;
  const _Float16* r = p + (size_t)n * ld;
  AF f;
  f.h2[0] = *(const v8h*)(r + 8 * g);
  f.h2[1] = *(const v8h*)(r + 16 + 8 * g);
  return f.v;
}

// A fragment from fp32 row-major source (converted to f16 at load).
__device__ inline v16h ldfrag_a_f32(const float* p, int ld) {
  int lane = threadIdx.x & 31;
  int m = lane & 15, g = lane >> 4;
  const float* r = p + (size_t)m * ld;
  AF f;
#pragma unroll
  for (int i = 0; i < 8; ++i) f.e[i] = (_Float16)r[8 * g + i];
#pragma unroll
  for (int i = 0; i < 8; ++i) f.e[8 + i] = (_Float16)r[16 + 8 * g + i];
  return f.v;
}

// ---------------- image [b][c][n] -> xT f16 [b][n][c] ----------------
__global__ void k_transpose_x(const float* __restrict__ img, _Float16* __restrict__ xT) {
  int idx = blockIdx.x * blockDim.x + threadIdx.x;
  if (idx >= B_ * C_ * N_) return;
  int n = idx & (N_ - 1);
  int c = (idx >> 10) & (C_ - 1);
  int b = idx >> 15;
  xT[((size_t)b * N_ + n) * C_ + c] = (_Float16)img[idx];
}

// ---- QKV projection (K=32 WMMA) with fused einops rearrange scatter ----
__global__ void __launch_bounds__(256) k_proj_qkv(
    const float* __restrict__ w_qkv, const float* __restrict__ b_qkv,
    const _Float16* __restrict__ xT,
    _Float16* __restrict__ qT, _Float16* __restrict__ kbuf, _Float16* __restrict__ vbuf) {
  int wave = threadIdx.x >> 5;
  int lane = threadIdx.x & 31;
  int b  = blockIdx.z;
  int n0 = blockIdx.x * 64;
  int o0 = blockIdx.y * 128 + wave * 16;           // 12288 output rows total
  const _Float16* xTb = xT + (size_t)b * N_ * C_;

  v16h a = ldfrag_a_f32(w_qkv + (size_t)o0 * C_, C_);
  v8f acc[4];
  v8f z = {0.f, 0.f, 0.f, 0.f, 0.f, 0.f, 0.f, 0.f};
#pragma unroll
  for (int j = 0; j < 4; ++j) {
    v16h bf = ldfrag_bt(xTb + (size_t)(n0 + 16 * j) * C_, C_);
    acc[j] = wmma16(a, bf, z);
  }

  int g = lane >> 4, nl = lane & 15;
  int sec = o0 >> 12;   // wave-uniform: 0=q 1=k 2=v (tiles never straddle sections)
#pragma unroll
  for (int j = 0; j < 4; ++j) {
    DF d; d.v = acc[j];
    int n = n0 + 16 * j + nl;
#pragma unroll
    for (int r = 0; r < 8; ++r) {
      int o  = o0 + r + 8 * g;
      int ch = o & 4095;
      int h  = ch & 3;                // channel = cd*HEADS + h
      int cd = ch >> 2;
      int bh = b * HEADS + h;
      _Float16 hv = (_Float16)(d.e[r] + b_qkv[o]);
      if (sec == 0)      qT  [((size_t)bh * N_ + n) * D_ + cd] = hv;
      else if (sec == 1) kbuf[((size_t)bh * D_ + cd) * N_ + n] = hv;
      else               vbuf[((size_t)bh * D_ + cd) * N_ + n] = hv;
    }
  }
}

// -------- row softmax over 1024 contiguous f16 elems; one wave per row --------
__global__ void __launch_bounds__(256) k_softmax_rows(_Float16* __restrict__ buf, float scale) {
  int wave = threadIdx.x >> 5, lane = threadIdx.x & 31;
  size_t row = (size_t)blockIdx.x * 8 + wave;       // 16384 rows
  _Float16* p = buf + row * 1024 + lane * 32;
  float x[32];
#pragma unroll
  for (int c = 0; c < 4; ++c) {
    HF t; t.v = *(const v8h*)(p + c * 8);
#pragma unroll
    for (int i = 0; i < 8; ++i) x[c * 8 + i] = (float)t.e[i];
  }
  float mx = -3.4e38f;
#pragma unroll
  for (int i = 0; i < 32; ++i) mx = fmaxf(mx, x[i]);
  for (int off = 16; off; off >>= 1) mx = fmaxf(mx, __shfl_xor(mx, off, 32));
  float s = 0.f;
#pragma unroll
  for (int i = 0; i < 32; ++i) { x[i] = __expf(x[i] - mx); s += x[i]; }
  for (int off = 16; off; off >>= 1) s += __shfl_xor(s, off, 32);
  float inv = scale / s;
#pragma unroll
  for (int c = 0; c < 4; ++c) {
    HF t;
#pragma unroll
    for (int i = 0; i < 8; ++i) t.e[i] = (_Float16)(x[c * 8 + i] * inv);
    *(v8h*)(p + c * 8) = t.v;
  }
}

// ---- batched NT GEMM: C[m,n] = sum_k A[m,k]*Bt[n,k]; 1024^3 f16 in, f32 acc ----
// Workgroup tile 128(M)x256(N), 8 waves of 64x64; K staged through LDS with
// double-buffered async global->LDS copies; 16 wmmas per wave per K=32 step.
__global__ void __launch_bounds__(256) k_gemm_nt(
    const _Float16* __restrict__ A, const _Float16* __restrict__ Bt,
    _Float16* __restrict__ Cmat) {
  __shared__ _Float16 sA[2][MT][KSTEP];    // 16 KB
  __shared__ _Float16 sB[2][NTL][KSTEP];   // 32 KB
  int tid  = threadIdx.x;
  int wave = tid >> 5, lane = tid & 31;
  int wm = wave >> 2, wn = wave & 3;       // 2 x 4 wave grid
  size_t base = (size_t)blockIdx.z * (1024u * 1024u);
  const _Float16* Ab = A  + base + (size_t)(blockIdx.y * MT)  * 1024;
  const _Float16* Bb = Bt + base + (size_t)(blockIdx.x * NTL) * 1024;
  _Float16* Cb = Cmat + base;

  v8f acc[4][4];
  v8f z = {0.f, 0.f, 0.f, 0.f, 0.f, 0.f, 0.f, 0.f};
#pragma unroll
  for (int i = 0; i < 4; ++i)
#pragma unroll
    for (int j = 0; j < 4; ++j) acc[i][j] = z;

  auto stage = [&](int buf, int k) {
#pragma unroll
    for (int j = 0; j < 2; ++j) {        // A: 512 x 16B chunks
      int c = tid + 256 * j;
      int row = c >> 2, col = (c & 3) * 8;
      async_cp16(Ab + (size_t)row * 1024 + k + col, &sA[buf][row][col]);
    }
#pragma unroll
    for (int j = 0; j < 4; ++j) {        // B: 1024 x 16B chunks
      int c = tid + 256 * j;
      int row = c >> 2, col = (c & 3) * 8;
      async_cp16(Bb + (size_t)row * 1024 + k + col, &sB[buf][row][col]);
    }
  };

  stage(0, 0);
  async_wait0();
  __syncthreads();

  int m0 = wm * 64, n0 = wn * 64;
  for (int step = 0; step < 1024 / KSTEP; ++step) {
    int cur = step & 1;
    if (step + 1 < 1024 / KSTEP) stage(cur ^ 1, (step + 1) * KSTEP);

    v16h af[4], bf[4];
#pragma unroll
    for (int i = 0; i < 4; ++i) af[i] = ldfrag_a(&sA[cur][m0 + 16 * i][0], KSTEP);
#pragma unroll
    for (int i = 0; i < 4; ++i) bf[i] = ldfrag_bt(&sB[cur][n0 + 16 * i][0], KSTEP);
#pragma unroll
    for (int i = 0; i < 4; ++i)
#pragma unroll
      for (int j = 0; j < 4; ++j) acc[i][j] = wmma16(af[i], bf[j], acc[i][j]);

    async_wait0();
    __syncthreads();
  }

  int g = lane >> 4, nl = lane & 15;
  int mbase = blockIdx.y * MT  + m0;
  int nbase = blockIdx.x * NTL + n0;
#pragma unroll
  for (int mi = 0; mi < 4; ++mi)
#pragma unroll
    for (int ni = 0; ni < 4; ++ni) {
      DF d; d.v = acc[mi][ni];
#pragma unroll
      for (int r = 0; r < 8; ++r)
        Cb[(size_t)(mbase + 16 * mi + r + 8 * g) * 1024 + (nbase + 16 * ni + nl)] =
            (_Float16)d.e[r];
    }
}

__global__ void k_zero(float* __restrict__ p, int count) {
  int i = blockIdx.x * blockDim.x + threadIdx.x;
  if (i < count) p[i] = 0.f;
}

// ---- thin output projection: proj[b][o][n] += sum_ch w_out[o][ch]*attn[b][ch][n] ----
__global__ void __launch_bounds__(256) k_proj_out(
    const float* __restrict__ w_out, const float* __restrict__ b_out,
    const _Float16* __restrict__ attn, float* __restrict__ proj) {
  int b   = blockIdx.z;
  int ch0 = blockIdx.y * 512;
  int n   = blockIdx.x * 256 + threadIdx.x;
  const _Float16* ap = attn + ((size_t)b * HID + ch0) * N_ + n;
  float acc[32];
#pragma unroll
  for (int o = 0; o < 32; ++o) acc[o] = 0.f;
  for (int cc = 0; cc < 512; ++cc) {
    float a = (float)ap[(size_t)cc * N_];
    const float* wr = w_out + (ch0 + cc);          // lane-uniform -> s_loads
#pragma unroll
    for (int o = 0; o < 32; ++o) acc[o] = fmaf(wr[(size_t)o * HID], a, acc[o]);
  }
#pragma unroll
  for (int o = 0; o < 32; ++o) {
    float add = acc[o] + (blockIdx.y == 0 ? b_out[o] : 0.f);
    atomicAdd(&proj[((size_t)b * C_ + o) * N_ + n], add);
  }
}

// ---- per-batch sum / sumsq for GroupNorm(1, C); one block per batch ----
__global__ void __launch_bounds__(256) k_stats(const float* __restrict__ proj,
                                               float* __restrict__ stats) {
  int b = blockIdx.x;
  const float* p = proj + (size_t)b * C_ * N_;
  float s = 0.f, s2 = 0.f;
  for (int i = threadIdx.x; i < C_ * N_; i += 256) {
    float v = p[i]; s += v; s2 += v * v;
  }
  __shared__ float sh0[256], sh1[256];
  sh0[threadIdx.x] = s; sh1[threadIdx.x] = s2;
  __syncthreads();
  for (int off = 128; off; off >>= 1) {
    if ((int)threadIdx.x < off) {
      sh0[threadIdx.x] += sh0[threadIdx.x + off];
      sh1[threadIdx.x] += sh1[threadIdx.x + off];
    }
    __syncthreads();
  }
  if (threadIdx.x == 0) { stats[b * 2] = sh0[0]; stats[b * 2 + 1] = sh1[0]; }
}

__global__ void k_gn(const float* __restrict__ proj, const float* __restrict__ stats,
                     const float* __restrict__ gamma, const float* __restrict__ beta,
                     float* __restrict__ out) {
  int idx = blockIdx.x * blockDim.x + threadIdx.x;   // B*C*N = 131072
  if (idx >= B_ * C_ * N_) return;
  int b = idx >> 15;
  int c = (idx >> 10) & 31;
  float inv_cnt = 1.0f / (float)(C_ * N_);
  float mean = stats[b * 2] * inv_cnt;
  float var  = stats[b * 2 + 1] * inv_cnt - mean * mean;
  float v = (proj[idx] - mean) * rsqrtf(var + 1e-5f);
  out[idx] = v * gamma[c] + beta[c];
}

extern "C" void kernel_launch(void* const* d_in, const int* in_sizes, int n_in,
                              void* d_out, int out_size, void* d_ws, size_t ws_size,
                              hipStream_t stream) {
  const float* image = (const float*)d_in[0];
  const float* w_qkv = (const float*)d_in[1];
  const float* b_qkv = (const float*)d_in[2];
  const float* w_out = (const float*)d_in[3];
  const float* b_out = (const float*)d_in[4];
  const float* gamma = (const float*)d_in[5];
  const float* beta  = (const float*)d_in[6];
  float* out = (float*)d_out;

  char* ws = (char*)d_ws;
  size_t off = 0;
  auto alloc = [&](size_t bytes) {
    char* p = ws + off;
    off += (bytes + 255) & ~(size_t)255;
    return p;
  };
  const size_t MAT = (size_t)BH * N_ * D_ * sizeof(_Float16);  // 32 MB each
  _Float16* xT   = (_Float16*)alloc((size_t)B_ * N_ * C_ * sizeof(_Float16));
  _Float16* qT   = (_Float16*)alloc(MAT);
  _Float16* kbuf = (_Float16*)alloc(MAT);
  _Float16* vbuf = (_Float16*)alloc(MAT);
  _Float16* ctxT = (_Float16*)alloc(MAT);
  _Float16* attn = (_Float16*)alloc(MAT);
  float* proj  = (float*)alloc((size_t)B_ * C_ * N_ * sizeof(float));
  float* stats = (float*)alloc(64);
  (void)in_sizes; (void)n_in; (void)out_size; (void)ws_size;

  // 1) image -> xT f16 [b][n][c]
  k_transpose_x<<<(B_ * C_ * N_ + 255) / 256, 256, 0, stream>>>(image, xT);
  // 2) QKV projection (WMMA, K=32) + rearrange scatter
  k_proj_qkv<<<dim3(N_ / 64, 12288 / 128, B_), 256, 0, stream>>>(
      w_qkv, b_qkv, xT, qT, kbuf, vbuf);
  // 3) softmaxes: q over feature dim (rows of qT, *DIM_HEAD^-0.5), k over spatial dim
  k_softmax_rows<<<2048, 256, 0, stream>>>(qT, 0.17677669529663687f);
  k_softmax_rows<<<2048, 256, 0, stream>>>(kbuf, 1.0f);
  // 4) ctxT[e,d] = sum_n V[e,n]*K[d,n]   (batched over 16 (b,h))
  k_gemm_nt<<<dim3(N_ / NTL, D_ / MT, BH), 256, 0, stream>>>(vbuf, kbuf, ctxT);
  // 5) attn[e,n] = sum_d ctxT[e,d]*qT[n,d]
  k_gemm_nt<<<dim3(N_ / NTL, D_ / MT, BH), 256, 0, stream>>>(ctxT, qT, attn);
  // 6) output projection (atomic accumulate over ch chunks)
  k_zero<<<(B_ * C_ * N_ + 255) / 256, 256, 0, stream>>>(proj, B_ * C_ * N_);
  k_proj_out<<<dim3(N_ / 256, HID / 512, B_), 256, 0, stream>>>(w_out, b_out, attn, proj);
  // 7) GroupNorm(1, C)
  k_stats<<<B_, 256, 0, stream>>>(proj, stats);
  k_gn<<<(B_ * C_ * N_ + 255) / 256, 256, 0, stream>>>(proj, stats, gamma, beta, out);
}
